// MultiheadAttention_17437567222132
// MI455X (gfx1250) — matrix operand order repaired and checked by reading, not compiled, not run on hardware
//
#include <hip/hip_runtime.h>
#include <hip/hip_bf16.h>
#include <stdint.h>

typedef __attribute__((ext_vector_type(16))) __bf16 v16bf;
typedef __attribute__((ext_vector_type(8)))  float  v8f;

// Problem dims
static constexpr int Bc  = 2;
static constexpr int Tc  = 2048;
static constexpr int Dc  = 1024;
static constexpr int Hc  = 16;
static constexpr int HDc = 64;
static constexpr int N3D = 3 * Dc;          // 3072
static constexpr int BT  = Bc * Tc;         // 4096

__device__ __forceinline__ unsigned short f2bf(float f) {
  unsigned u = __builtin_bit_cast(unsigned, f);
  u += 0x7fffu + ((u >> 16) & 1u);          // round-to-nearest-even
  return (unsigned short)(u >> 16);
}
__device__ __forceinline__ __bf16 bfraw(unsigned short u) {
  return __builtin_bit_cast(__bf16, u);
}
__device__ __forceinline__ v8f wmma_bf16(v16bf a, v16bf b, v8f c) {
  return __builtin_amdgcn_wmma_f32_16x16x32_bf16(
      /*neg_a=*/false, a, /*neg_b=*/false, b,
      /*c_mod=*/(short)0, c, /*reuse_a=*/false, /*reuse_b=*/false);
}

// A fragment: 16x32 bf16, row-major source, row = lane&15.
// ISA layout: lanes 0-15 hold K = {0..7, 16..23}; lanes 16-31 hold K = {8..15, 24..31}.
__device__ __forceinline__ v16bf load_frag_a(const unsigned short* p, int ld) {
  const int lane = threadIdx.x & 31;
  const unsigned short* rp = p + (size_t)(lane & 15) * ld + ((lane >> 4) << 3);
  v16bf f;
#pragma unroll
  for (int j = 0; j < 8; ++j) {
    int k = ((j >> 2) << 4) + ((j & 3) << 1);
    unsigned u = *(const unsigned*)(rp + k);
    f[2 * j]     = bfraw((unsigned short)(u & 0xffffu));
    f[2 * j + 1] = bfraw((unsigned short)(u >> 16));
  }
  return f;
}

// B fragment: 32x16 bf16, sourced as N-major rows (row = N = lane&15, K contiguous).
// ISA layout: lanes 0-15 hold K = {0..7, 16..23}; lanes 16-31 hold K = {8..15, 24..31}.
__device__ __forceinline__ v16bf load_frag_b(const unsigned short* p, int ld) {
  const int lane = threadIdx.x & 31;
  const unsigned short* rp = p + (size_t)(lane & 15) * ld + ((lane >> 4) << 3);
  v16bf f;
#pragma unroll
  for (int j = 0; j < 8; ++j) {
    unsigned u = *(const unsigned*)(rp + ((j >> 2) << 4) + ((j & 3) << 1));
    f[2 * j]     = bfraw((unsigned short)(u & 0xffffu));
    f[2 * j + 1] = bfraw((unsigned short)(u >> 16));
  }
  return f;
}

// ---------------- conversion kernels ----------------
__global__ void mha_cvt_x(const float* __restrict__ x, unsigned short* __restrict__ xh, int n) {
  int i = blockIdx.x * blockDim.x + threadIdx.x;
  if (i < n) xh[i] = f2bf(x[i]);
}
// Wt[n*rows + k] = W[k*cols + n]   (store transposed so B fragments are K-contiguous)
__global__ void mha_cvt_wT(const float* __restrict__ W, unsigned short* __restrict__ Wt,
                           int rows /*K*/, int cols /*N*/) {
  int i = blockIdx.x * blockDim.x + threadIdx.x;
  if (i < rows * cols) {
    int nn = i / rows, k = i % rows;
    Wt[i] = f2bf(W[(size_t)k * cols + nn]);
  }
}

// Scatter one 16x16 QKV result tile (bias added) into q / k / vT (bf16).
__device__ __forceinline__ void store_qkv_tile(
    v8f c, int m0, int n0, const float* __restrict__ b_qkv,
    unsigned short* __restrict__ qb, unsigned short* __restrict__ kb,
    unsigned short* __restrict__ vTb) {
  const int lane = threadIdx.x & 31;
  const int hf = lane >> 4, n = lane & 15;
  const int nn = n0 + n;
  const int h = nn / 192, rem = nn % 192, kind = rem >> 6, d = rem & 63;
  const float bias = b_qkv[nn];
#pragma unroll
  for (int j = 0; j < 8; ++j) {
    int m = m0 + j + (hf << 3);
    int bb = m >> 11, t = m & 2047;
    unsigned short v = f2bf(c[j] + bias);
    size_t bh = (size_t)(bb * Hc + h);
    if (kind == 0)      qb [(bh * Tc + t) * HDc + d] = v;
    else if (kind == 1) kb [(bh * Tc + t) * HDc + d] = v;
    else                vTb[(bh * HDc + d) * Tc + t] = v;   // v stored transposed
  }
}

// ---------------- QKV GEMM: (4096 x 1024) x (1024 x 3072) ----------------
// 2x2 register blocking: each wave owns a 32x32 macro-tile (4 WMMAs / 4 fragment
// loads per k-step -> 2x operand reuse vs single-tile waves).
__global__ __launch_bounds__(256) void mha_qkv_gemm(
    const unsigned short* __restrict__ xh, const unsigned short* __restrict__ WqkvT,
    const float* __restrict__ b_qkv,
    unsigned short* __restrict__ qb, unsigned short* __restrict__ kb,
    unsigned short* __restrict__ vTb) {
  const int wave = threadIdx.x >> 5;
  const int tile = blockIdx.x * 8 + wave;          // (4096/32)*(3072/32) = 12288 tiles
  const int m0 = (tile / 96) << 5;
  const int n0 = (tile % 96) << 5;
  const unsigned short* ap0 = xh    + (size_t)m0 * Dc;
  const unsigned short* ap1 = ap0   + (size_t)16 * Dc;
  const unsigned short* bp0 = WqkvT + (size_t)n0 * Dc;
  const unsigned short* bp1 = bp0   + (size_t)16 * Dc;
  v8f c00 = {}, c01 = {}, c10 = {}, c11 = {};
#pragma unroll 2
  for (int kt = 0; kt < Dc; kt += 32) {
    __builtin_prefetch(ap0 + kt + 128, 0, 1);      // -> global_prefetch_b8
    v16bf a0 = load_frag_a(ap0 + kt, Dc);
    v16bf a1 = load_frag_a(ap1 + kt, Dc);
    v16bf b0 = load_frag_b(bp0 + kt, Dc);
    v16bf b1 = load_frag_b(bp1 + kt, Dc);
    c00 = wmma_bf16(a0, b0, c00);
    c01 = wmma_bf16(a0, b1, c01);
    c10 = wmma_bf16(a1, b0, c10);
    c11 = wmma_bf16(a1, b1, c11);
  }
  store_qkv_tile(c00, m0,      n0,      b_qkv, qb, kb, vTb);
  store_qkv_tile(c01, m0,      n0 + 16, b_qkv, qb, kb, vTb);
  store_qkv_tile(c10, m0 + 16, n0,      b_qkv, qb, kb, vTb);
  store_qkv_tile(c11, m0 + 16, n0 + 16, b_qkv, qb, kb, vTb);
}

// ---------------- fused scores -> softmax -> AV, per (b,h,16-query tile) ----------------
__global__ __launch_bounds__(128) void mha_attn(
    const unsigned short* __restrict__ qb, const unsigned short* __restrict__ kb,
    const unsigned short* __restrict__ vTb,
    float* __restrict__ attn_out, unsigned short* __restrict__ vals) {
  extern __shared__ float s[];                     // [16][2048] f32 scores = 128 KB
  const int wave = threadIdx.x >> 5;
  const int lane = threadIdx.x & 31;
  const int hf = lane >> 4, n = lane & 15;
  const int bh = blockIdx.x >> 7;                  // b*H + h  (32)
  const int qt = blockIdx.x & 127;                 // query tile (128)
  const unsigned short* qp = qb  + ((size_t)bh * Tc + qt * 16) * HDc;
  const unsigned short* kp = kb  + (size_t)bh * Tc * HDc;
  const unsigned short* vp = vTb + (size_t)bh * HDc * Tc;

  // Phase 1: scores = q . k^T * 1/sqrt(64). q fragments hoisted (reused 32x per wave).
  v16bf a0 = load_frag_a(qp + 0,  HDc);
  v16bf a1 = load_frag_a(qp + 32, HDc);
  for (int jt = wave; jt < 128; jt += 4) {
    v16bf b0 = load_frag_b(kp + (size_t)jt * 16 * HDc + 0,  HDc);
    v16bf b1 = load_frag_b(kp + (size_t)jt * 16 * HDc + 32, HDc);
    v8f c = {};
    c = wmma_bf16(a0, b0, c);
    c = wmma_bf16(a1, b1, c);
#pragma unroll
    for (int j = 0; j < 8; ++j)
      s[(size_t)(j + (hf << 3)) * Tc + jt * 16 + n] = c[j] * 0.125f;
  }
  __syncthreads();

  // Phase 2: row softmax (wave32 shuffle reductions), write attn to global.
  for (int r = wave * 4; r < wave * 4 + 4; ++r) {
    float* row = s + (size_t)r * Tc;
    float mx = -3.4e38f;
    for (int j = lane; j < Tc; j += 32) mx = fmaxf(mx, row[j]);
#pragma unroll
    for (int o = 16; o > 0; o >>= 1) mx = fmaxf(mx, __shfl_xor(mx, o, 32));
    float sum = 0.f;
    for (int j = lane; j < Tc; j += 32) { float e = __expf(row[j] - mx); row[j] = e; sum += e; }
#pragma unroll
    for (int o = 16; o > 0; o >>= 1) sum += __shfl_xor(sum, o, 32);
    float inv = 1.f / sum;
    float* g = attn_out + ((size_t)bh * Tc + qt * 16 + r) * Tc;
    for (int j = lane; j < Tc; j += 32) { float v = row[j] * inv; row[j] = v; g[j] = v; }
  }
  __syncthreads();

  // Phase 3: values = attn x v. Each wave owns one 16-wide slice of HD=64.
  const int n0 = wave << 4;
  v8f c = {};
  for (int kt = 0; kt < Tc; kt += 32) {
    const float* srow = s + (size_t)(lane & 15) * Tc + kt + (hf << 3);
    v16bf a;
#pragma unroll
    for (int j = 0; j < 8; ++j) {
      int k = ((j >> 2) << 4) + ((j & 3) << 1);
      a[2 * j]     = bfraw(f2bf(srow[k]));
      a[2 * j + 1] = bfraw(f2bf(srow[k + 1]));
    }
    v16bf b = load_frag_b(vp + (size_t)n0 * Tc + kt, Tc);  // vT rows are K-contiguous
    c = wmma_bf16(a, b, c);
  }
  const int bb = bh >> 4, h = bh & 15;
#pragma unroll
  for (int j = 0; j < 8; ++j) {
    int t = qt * 16 + j + (hf << 3);
    vals[(size_t)(bb * Tc + t) * Dc + h * HDc + n0 + n] = f2bf(c[j]);
  }
}

// ---------------- output projection: (4096 x 1024) x (1024 x 1024) + bias ----------------
// Same 2x2 register blocking as the QKV GEMM.
__global__ __launch_bounds__(256) void mha_outproj(
    const unsigned short* __restrict__ vals, const unsigned short* __restrict__ WoutT,
    const float* __restrict__ b_out, float* __restrict__ out) {
  const int wave = threadIdx.x >> 5;
  const int lane = threadIdx.x & 31;
  const int tile = blockIdx.x * 8 + wave;          // (4096/32)*(1024/32) = 4096 tiles
  const int m0 = (tile >> 5) << 5;
  const int n0 = (tile & 31) << 5;
  const unsigned short* ap0 = vals  + (size_t)m0 * Dc;
  const unsigned short* ap1 = ap0   + (size_t)16 * Dc;
  const unsigned short* bp0 = WoutT + (size_t)n0 * Dc;
  const unsigned short* bp1 = bp0   + (size_t)16 * Dc;
  v8f c00 = {}, c01 = {}, c10 = {}, c11 = {};
#pragma unroll 2
  for (int kt = 0; kt < Dc; kt += 32) {
    __builtin_prefetch(ap0 + kt + 128, 0, 1);
    v16bf a0 = load_frag_a(ap0 + kt, Dc);
    v16bf a1 = load_frag_a(ap1 + kt, Dc);
    v16bf b0 = load_frag_b(bp0 + kt, Dc);
    v16bf b1 = load_frag_b(bp1 + kt, Dc);
    c00 = wmma_bf16(a0, b0, c00);
    c01 = wmma_bf16(a0, b1, c01);
    c10 = wmma_bf16(a1, b0, c10);
    c11 = wmma_bf16(a1, b1, c11);
  }
  const int hf = lane >> 4, n = lane & 15;
  const float bias0 = b_out[n0 + n];
  const float bias1 = b_out[n0 + 16 + n];
#pragma unroll
  for (int j = 0; j < 8; ++j) {
    size_t r0 = (size_t)(m0 + j + (hf << 3)) * Dc;
    size_t r1 = (size_t)(m0 + 16 + j + (hf << 3)) * Dc;
    out[r0 + n0 + n]      = c00[j] + bias0;
    out[r0 + n0 + 16 + n] = c01[j] + bias1;
    out[r1 + n0 + n]      = c10[j] + bias0;
    out[r1 + n0 + 16 + n] = c11[j] + bias1;
  }
}

extern "C" void kernel_launch(void* const* d_in, const int* in_sizes, int n_in,
                              void* d_out, int out_size, void* d_ws, size_t ws_size,
                              hipStream_t stream) {
  const float* x     = (const float*)d_in[0];
  const float* W_qkv = (const float*)d_in[1];
  const float* b_qkv = (const float*)d_in[2];
  const float* W_out = (const float*)d_in[3];
  const float* b_out = (const float*)d_in[4];

  float* out  = (float*)d_out;                         // [B,T,D] = 4,194,304 f32
  float* attn = out + (size_t)BT * Dc;                 // [B,H,T,T]

  // Workspace layout (u16 elements), ~48 MB total.
  unsigned short* ws = (unsigned short*)d_ws;
  size_t o = 0;
  unsigned short* xh    = ws + o; o += (size_t)BT * Dc;        // 4M
  unsigned short* WqkvT = ws + o; o += (size_t)N3D * Dc;       // 3M
  unsigned short* WoutT = ws + o; o += (size_t)Dc * Dc;        // 1M
  unsigned short* qb    = ws + o; o += (size_t)Bc * Hc * Tc * HDc;  // 4M
  unsigned short* kb    = ws + o; o += (size_t)Bc * Hc * Tc * HDc;  // 4M
  unsigned short* vTb   = ws + o; o += (size_t)Bc * Hc * HDc * Tc;  // 4M
  unsigned short* vals  = ws + o; o += (size_t)BT * Dc;        // 4M

  mha_cvt_x<<<(BT * Dc + 255) / 256, 256, 0, stream>>>(x, xh, BT * Dc);
  mha_cvt_wT<<<(N3D * Dc + 255) / 256, 256, 0, stream>>>(W_qkv, WqkvT, Dc, N3D);
  mha_cvt_wT<<<(Dc * Dc + 255) / 256, 256, 0, stream>>>(W_out, WoutT, Dc, Dc);

  mha_qkv_gemm<<<(128 * 96) / 8, 256, 0, stream>>>(xh, WqkvT, b_qkv, qb, kb, vTb);

  (void)hipFuncSetAttribute(reinterpret_cast<const void*>(mha_attn),
                            hipFuncAttributeMaxDynamicSharedMemorySize, 16 * Tc * 4);
  mha_attn<<<Bc * Hc * (Tc / 16), 128, 16 * Tc * 4, stream>>>(qb, kb, vTb, attn, vals);

  mha_outproj<<<(128 * 32) / 8, 256, 0, stream>>>(vals, WoutT, b_out, out);
}
